// SimpleHybridQNN_10428180595412
// MI455X (gfx1250) — matrix operand order repaired and checked by reading, not compiled, not run on hardware
//
#include <hip/hip_runtime.h>
#include <hip/hip_bf16.h>
#include <math.h>

// ---------------------------------------------------------------------------
// SimpleHybridQNN forward, MI455X (gfx1250, wave32).
// All GEMM layers run through v_wmma_f32_16x16x32_f16 (f16 in / f32 acc),
// K-loop ping-pong double-buffered (unroll-2, no rotation copies),
// 16x64 (or 16x32) tile per wave. Workspace budget ~850 MB (aliased regions).
// ---------------------------------------------------------------------------

#define BATCH 16384

typedef _Float16 h16;
typedef __attribute__((ext_vector_type(16))) _Float16 v16h;
typedef __attribute__((ext_vector_type(8)))  _Float16 v8h;
typedef __attribute__((ext_vector_type(8)))  float    v8f;

// ---------------------------------------------------------------------------
// Pack weights (f32 [N][K] row-major, or conv OIHW) into WMMA B-fragment
// layout: [nt][kc][lane][16 halves], lane<16 -> n=nt*16+lane, halves = K
// kc*32+{0..7,16..23} ; lane>=16 -> same n, halves = K kc*32+{8..15,24..31}. Zero-pads K tail.
// ---------------------------------------------------------------------------
__global__ void pack_weights(const float* __restrict__ W, h16* __restrict__ out,
                             int N, int Kreal, int Kp, int Cin, int total) {
  int tid = blockIdx.x * blockDim.x + threadIdx.x;
  if (tid >= total) return;                       // total = N * Kp
  int h    = tid & 15;
  int lane = (tid >> 4) & 31;
  int kch  = Kp >> 5;
  int kc   = (tid >> 9) % kch;
  int nt   = tid / (kch << 9);
  int n    = (nt << 4) + (lane & 15);
  int k    = (kc << 5) + ((h < 8) ? h : (h + 8)) + ((lane < 16) ? 0 : 8);
  float v = 0.f;
  if (k < Kreal) {
    if (Cin == 0) {
      v = W[(size_t)n * Kreal + k];               // linear: W[n][k]
    } else {                                      // conv OIHW, k = pos*Cin + ic
      int pos = k / Cin, ic = k % Cin;
      v = W[((size_t)n * Cin + ic) * 9 + pos];
    }
  }
  out[tid] = (h16)v;
}

// ---------------------------------------------------------------------------
// conv1: 1->16ch, 5x5, stride 2, pad 2, 28x28 -> 14x14, +bias, ReLU, BN.
// Output NHWC f16 [B][14][14][16].
// ---------------------------------------------------------------------------
__global__ void conv1_bn_relu(const float* __restrict__ x,
                              const float* __restrict__ w,
                              const float* __restrict__ bias,
                              const float* __restrict__ g,  const float* __restrict__ be,
                              const float* __restrict__ mu, const float* __restrict__ var,
                              h16* __restrict__ out, int total) {
  int tid = blockIdx.x * blockDim.x + threadIdx.x;
  if (tid >= total) return;                       // total = B*196*16
  int oc  = tid & 15;
  int pix = (tid >> 4) % 196;
  int b   = tid / (196 * 16);
  int oy = pix / 14, ox = pix % 14;
  const float* xi = x + (size_t)b * 784;
  const float* wk = w + oc * 25;
  float acc = bias[oc];
  #pragma unroll
  for (int ky = 0; ky < 5; ++ky) {
    int iy = oy * 2 + ky - 2;
    if ((unsigned)iy >= 28u) continue;
    #pragma unroll
    for (int kx = 0; kx < 5; ++kx) {
      int ix = ox * 2 + kx - 2;
      if ((unsigned)ix >= 28u) continue;
      acc += xi[iy * 28 + ix] * wk[ky * 5 + kx];
    }
  }
  acc = fmaxf(acc, 0.f);
  float s = g[oc] * rsqrtf(var[oc] + 1e-5f);
  acc = (acc - mu[oc]) * s + be[oc];
  out[tid] = (h16)acc;
}

// ---------------------------------------------------------------------------
// im2col for 3x3 convs on HxH NHWC input -> rows [B*49][Kp], k = pos*Cin+ic.
// ---------------------------------------------------------------------------
__global__ void im2col_3x3(const h16* __restrict__ in, h16* __restrict__ out,
                           int Hin, int Cin, int stride, int Kreal, int Kp, int total) {
  int tid = blockIdx.x * blockDim.x + threadIdx.x;
  if (tid >= total) return;
  int col = tid % Kp;
  int row = tid / Kp;
  int pix = row % 49;
  int b   = row / 49;
  int oy = pix / 7, ox = pix % 7;
  h16 v = (h16)0.f;
  if (col < Kreal) {
    int pos = col / Cin, ic = col % Cin;
    int ky = pos / 3, kx = pos % 3;
    int iy = oy * stride - 1 + ky;
    int ix = ox * stride - 1 + kx;
    if ((unsigned)iy < (unsigned)Hin && (unsigned)ix < (unsigned)Hin)
      v = in[(((size_t)b * Hin + iy) * Hin + ix) * Cin + ic];
  }
  out[tid] = v;
}

// ---------------------------------------------------------------------------
// Generic WMMA GEMM: out[M][ldOut] (cols [0,N)) = epi(A[M][Kp] @ Wpacked + b)
// One wave computes a 16(M) x 16*NT(N) tile: one A fragment feeds NT WMMAs.
// K-loop is unrolled by 2 with ping-pong buffers: while buffer P's WMMAs
// issue, the loads refilling buffer 1-P are in flight (no rotation copies).
// act: 0=none 1=relu 2=tanh ; optional BN (g,b,m,v) applied AFTER act.
// ---------------------------------------------------------------------------
template <int NT>
__global__ __launch_bounds__(128) void gemm_wmma(
    const h16* __restrict__ A, const h16* __restrict__ Wp,
    const float* __restrict__ bias, h16* __restrict__ out,
    int M, int Kp, int N, int ldOut, int act,
    const float* __restrict__ bng, const float* __restrict__ bnb,
    const float* __restrict__ bnm, const float* __restrict__ bnv) {
  const int lane = threadIdx.x & 31;
  const int wave = blockIdx.x * (blockDim.x >> 5) + (threadIdx.x >> 5);
  const int nstrips = N / (16 * NT);
  const int ns = wave % nstrips;
  const int mt = wave / nstrips;
  if (mt >= (M >> 4)) return;                     // wave-uniform guard
  const int kchunks = Kp >> 5;
  const int laneHi  = lane >> 4;
  const int m = (mt << 4) + (lane & 15);

  const h16* arow = A + (size_t)m * Kp + (laneHi ? 8 : 0);
  const v16h* bp[NT];
  #pragma unroll
  for (int t = 0; t < NT; ++t)
    bp[t] = (const v16h*)Wp + (size_t)(ns * NT + t) * kchunks * 32 + lane;

  v8f acc[NT];
  #pragma unroll
  for (int t = 0; t < NT; ++t)
    #pragma unroll
    for (int i = 0; i < 8; ++i) acc[t][i] = 0.f;

  // A fragment (16x32 f16): lane<16 holds K {0..7,16..23}, lane>=16 {8..15,24..31};
  // two 16-B loads land directly in the low/high halves of the v16h operand.
  auto load_chunk = [&](int kc, v16h& a, v16h (&bb)[NT]) {
    ((v8h*)&a)[0] = *(const v8h*)(arow + (size_t)kc * 32);
    ((v8h*)&a)[1] = *(const v8h*)(arow + (size_t)kc * 32 + 16);
    #pragma unroll
    for (int t = 0; t < NT; ++t) bb[t] = bp[t][(size_t)kc * 32];
  };
  auto do_wmma = [&](const v16h& a, const v16h (&bb)[NT]) {
    #pragma unroll
    for (int t = 0; t < NT; ++t)
      acc[t] = __builtin_amdgcn_wmma_f32_16x16x32_f16(false, a, false, bb[t],
                                                      (short)0, acc[t], false, false);
  };

  v16h a0, a1;
  v16h b0[NT], b1[NT];
  load_chunk(0, a0, b0);
  int kc = 0;
  for (; kc + 2 < kchunks; kc += 2) {             // ping-pong main loop
    __builtin_prefetch(arow + (size_t)(kc + 8) * 32, 0, 3);   // speculative
    load_chunk(kc + 1, a1, b1);
    do_wmma(a0, b0);
    load_chunk(kc + 2, a0, b0);
    do_wmma(a1, b1);
  }
  if (kc + 1 < kchunks) {                         // 2 chunks remain
    load_chunk(kc + 1, a1, b1);
    do_wmma(a0, b0);
    do_wmma(a1, b1);
  } else {                                        // 1 chunk remains
    do_wmma(a0, b0);
  }

  // ---- epilogue: bias -> act -> (optional) BN, store f16 ----
  #pragma unroll
  for (int t = 0; t < NT; ++t) {
    const int n = ns * NT * 16 + t * 16 + (lane & 15);
    float bs = bias[n];
    float sc = 1.f, sh = 0.f;
    if (bng) {
      sc = bng[n] * rsqrtf(bnv[n] + 1e-5f);
      sh = bnb[n] - bnm[n] * sc;
    }
    #pragma unroll
    for (int r = 0; r < 8; ++r) {
      int mr = (mt << 4) + r + (laneHi << 3);     // C/D layout: vgpr r -> M=r (+8 hi)
      float v = acc[t][r] + bs;
      if (act == 1)      v = fmaxf(v, 0.f);
      else if (act == 2) v = tanhf(v);
      v = v * sc + sh;
      out[(size_t)mr * ldOut + n] = (h16)v;
    }
  }
}

// ---------------------------------------------------------------------------
// Overlapping 7x7 -> 2x2 mean pool; output flattened in NCHW order
// (k = c*4 + y*2 + x) to match the reference fc1 weight layout.
// ---------------------------------------------------------------------------
__global__ void pool7to2(const h16* __restrict__ in, h16* __restrict__ out, int total) {
  int tid = blockIdx.x * blockDim.x + threadIdx.x;
  if (tid >= total) return;                       // total = B*256
  int x = tid & 1, y = (tid >> 1) & 1, c = (tid >> 2) & 63, b = tid >> 8;
  const h16* base = in + (size_t)b * 49 * 64;
  int y0 = y * 3, x0 = x * 3;                     // windows [0..3] and [3..6]
  float s = 0.f;
  #pragma unroll
  for (int i = 0; i < 4; ++i)
    #pragma unroll
    for (int j = 0; j < 4; ++j)
      s += (float)base[((y0 + i) * 7 + (x0 + j)) * 64 + c];
  out[tid] = (h16)(s * (1.f / 16.f));
}

// x (f32 [B][784]) -> f16 [B][800] zero-padded for the q1 GEMM
__global__ void cast_pad_x(const float* __restrict__ x, h16* __restrict__ out, int total) {
  int tid = blockIdx.x * blockDim.x + threadIdx.x;
  if (tid >= total) return;                       // total = B*800
  int col = tid % 800, b = tid / 800;
  out[tid] = (col < 784) ? (h16)x[(size_t)b * 784 + col] : (h16)0.f;
}

// ---------------------------------------------------------------------------
// 8-qubit statevector sim, one wave32 per sample, 8 amplitudes per lane.
// idx = r*32 + lane : bits7..5 = r, bits4..0 = lane. Wire w acts on bit 7-w.
// Writes z-expectations into fcat cols 64..67 and zeros cols 68..95.
// ---------------------------------------------------------------------------
__global__ __launch_bounds__(256) void quantum_kernel(
    const h16* __restrict__ q, const float* __restrict__ qw,
    h16* __restrict__ fcat, int Bn) {
  const int lane   = threadIdx.x & 31;
  const int sample = blockIdx.x * 8 + (threadIdx.x >> 5);
  if (sample >= Bn) return;                       // wave-uniform
  float s[8];
  float sum = 0.f;
  #pragma unroll
  for (int r = 0; r < 8; ++r) {
    s[r] = (float)q[(size_t)sample * 256 + r * 32 + lane];
    sum += s[r] * s[r];
  }
  #pragma unroll
  for (int off = 16; off; off >>= 1) sum += __shfl_xor(sum, off, 32);
  float inv = 1.f / fmaxf(sqrtf(sum), 1e-12f);
  #pragma unroll
  for (int r = 0; r < 8; ++r) s[r] *= inv;

  for (int l = 0; l < 4; ++l) {
    // RY on each wire
    for (int i = 0; i < 8; ++i) {
      float th = qw[l * 8 + i] * 0.5f;
      float c = cosf(th), sn = sinf(th);
      int p = 7 - i;
      if (p >= 5) {                               // bit lives in r
        int qm = 1 << (p - 5);
        #pragma unroll
        for (int r = 0; r < 8; ++r)
          if (!(r & qm)) {
            float a0 = s[r], a1 = s[r | qm];
            s[r]      = c * a0 - sn * a1;
            s[r | qm] = sn * a0 + c * a1;
          }
      } else {                                    // bit lives in lane
        int bm = 1 << p;
        #pragma unroll
        for (int r = 0; r < 8; ++r) {
          float o = __shfl_xor(s[r], bm, 32);
          s[r] = (lane & bm) ? (sn * o + c * s[r]) : (c * s[r] - sn * o);
        }
      }
    }
    // CNOT chain: ctrl wire i (bit 7-i), target wire i+1 (bit 6-i)
    for (int i = 0; i < 7; ++i) {
      int pc = 7 - i, pt = 6 - i;
      if (pt >= 5) {                              // both bits in r
        int qc = 1 << (pc - 5), qt = 1 << (pt - 5);
        #pragma unroll
        for (int r = 0; r < 8; ++r)
          if ((r & qc) && !(r & qt)) {
            float t = s[r]; s[r] = s[r | qt]; s[r | qt] = t;
          }
      } else if (pc >= 5) {                       // ctrl in r, target in lane
        int qc = 1 << (pc - 5), bm = 1 << pt;
        #pragma unroll
        for (int r = 0; r < 8; ++r) {
          float o = __shfl_xor(s[r], bm, 32);
          if (r & qc) s[r] = o;
        }
      } else {                                    // both in lane
        int bmc = 1 << pc, bmt = 1 << pt;
        #pragma unroll
        for (int r = 0; r < 8; ++r) {
          float o = __shfl_xor(s[r], bmt, 32);
          if (lane & bmc) s[r] = o;
        }
      }
    }
  }
  // <Z> on wires 0..3 -> bits 7,6,5,4
  float z0 = 0.f, z1 = 0.f, z2 = 0.f, z3 = 0.f;
  #pragma unroll
  for (int r = 0; r < 8; ++r) {
    float pr = s[r] * s[r];
    z0 += (r & 4)     ? -pr : pr;
    z1 += (r & 2)     ? -pr : pr;
    z2 += (r & 1)     ? -pr : pr;
    z3 += (lane & 16) ? -pr : pr;
  }
  #pragma unroll
  for (int off = 16; off; off >>= 1) {
    z0 += __shfl_xor(z0, off, 32);
    z1 += __shfl_xor(z1, off, 32);
    z2 += __shfl_xor(z2, off, 32);
    z3 += __shfl_xor(z3, off, 32);
  }
  h16* row = fcat + (size_t)sample * 96;
  if (lane == 0) {
    row[64] = (h16)z0; row[65] = (h16)z1; row[66] = (h16)z2; row[67] = (h16)z3;
  }
  if (lane < 28) row[68 + lane] = (h16)0.f;       // zero-pad K 68..95
}

// final 32->1 dot + sigmoid, f32 output
__global__ void f4_sigmoid(const h16* __restrict__ in, const float* __restrict__ w,
                           const float* __restrict__ b, float* __restrict__ out, int Bn) {
  int i = blockIdx.x * blockDim.x + threadIdx.x;
  if (i >= Bn) return;
  float acc = b[0];
  #pragma unroll
  for (int k = 0; k < 32; ++k) acc += (float)in[(size_t)i * 32 + k] * w[k];
  out[i] = 1.f / (1.f + expf(-acc));
}

// ---------------------------------------------------------------------------
extern "C" void kernel_launch(void* const* d_in, const int* in_sizes, int n_in,
                              void* d_out, int out_size, void* d_ws, size_t ws_size,
                              hipStream_t stream) {
  (void)in_sizes; (void)n_in; (void)out_size; (void)ws_size;
  const float* x     = (const float*)d_in[0];
  const float* c1w   = (const float*)d_in[1];
  const float* c1b   = (const float*)d_in[2];
  const float* bn1g  = (const float*)d_in[3];
  const float* bn1b  = (const float*)d_in[4];
  const float* bn1m  = (const float*)d_in[5];
  const float* bn1v  = (const float*)d_in[6];
  const float* c2w   = (const float*)d_in[7];
  const float* c2b   = (const float*)d_in[8];
  const float* bn2g  = (const float*)d_in[9];
  const float* bn2b  = (const float*)d_in[10];
  const float* bn2m  = (const float*)d_in[11];
  const float* bn2v  = (const float*)d_in[12];
  const float* c3w   = (const float*)d_in[13];
  const float* c3b   = (const float*)d_in[14];
  const float* fc1w  = (const float*)d_in[15];
  const float* fc1b  = (const float*)d_in[16];
  const float* fc2w  = (const float*)d_in[17];
  const float* fc2b  = (const float*)d_in[18];
  const float* q1w   = (const float*)d_in[19];
  const float* q1b   = (const float*)d_in[20];
  const float* q2w   = (const float*)d_in[21];
  const float* q2b   = (const float*)d_in[22];
  const float* qwts  = (const float*)d_in[23];
  const float* f1w   = (const float*)d_in[24];
  const float* f1b   = (const float*)d_in[25];
  const float* fbn1g = (const float*)d_in[26];
  const float* fbn1b = (const float*)d_in[27];
  const float* fbn1m = (const float*)d_in[28];
  const float* fbn1v = (const float*)d_in[29];
  const float* f2w   = (const float*)d_in[30];
  const float* f2b   = (const float*)d_in[31];
  const float* fbn2g = (const float*)d_in[32];
  const float* fbn2b = (const float*)d_in[33];
  const float* fbn2m = (const float*)d_in[34];
  const float* fbn2v = (const float*)d_in[35];
  const float* f3w   = (const float*)d_in[36];
  const float* f3b   = (const float*)d_in[37];
  const float* f4w   = (const float*)d_in[38];
  const float* f4b   = (const float*)d_in[39];

  char* base = (char*)d_ws;
  const int M49 = BATCH * 49;                           // 802816
  // Region 0 (462.4 MB): h1 [B][14][14][16], later A3 im2col [B*49][288]
  h16* h1  = (h16*)(base);
  h16* A3  = (h16*)(base);
  // Region 1 (256.9 MB): A2 im2col [B*49][160], later h3 [B*49][64]
  h16* A2  = (h16*)(base + 462422016);
  h16* h3  = (h16*)(base + 462422016);
  // Region 2 (51.4 MB): h2 [B*49][32]
  h16* h2  = (h16*)(base + 462422016 + 256901120);
  size_t off = 462422016 + 256901120 + 51380224;
  auto salloc = [&](size_t bytes) -> h16* {
    h16* p = (h16*)(base + off);
    off += (bytes + 255) & ~(size_t)255;
    return p;
  };
  h16* pooled = salloc((size_t)BATCH * 256 * 2);
  h16* fc1o   = salloc((size_t)BATCH * 128 * 2);
  h16* xq     = salloc((size_t)BATCH * 800 * 2);
  h16* q1o    = salloc((size_t)BATCH * 512 * 2);
  h16* q2o    = salloc((size_t)BATCH * 256 * 2);
  h16* fcat   = salloc((size_t)BATCH * 96 * 2);
  h16* f1o    = salloc((size_t)BATCH * 128 * 2);
  h16* f2o    = salloc((size_t)BATCH * 64 * 2);
  h16* f3o    = salloc((size_t)BATCH * 32 * 2);
  h16* pw2    = salloc(32 * 160 * 2);
  h16* pw3    = salloc(64 * 288 * 2);
  h16* pwfc1  = salloc(128 * 256 * 2);
  h16* pwfc2  = salloc(64 * 128 * 2);
  h16* pwq1   = salloc(512 * 800 * 2);
  h16* pwq2   = salloc(256 * 512 * 2);
  h16* pwf1   = salloc(128 * 96 * 2);
  h16* pwf2   = salloc(64 * 128 * 2);
  h16* pwf3   = salloc(32 * 64 * 2);

  auto pack = [&](const float* W, h16* dst, int N, int Kreal, int Kp, int Cin) {
    int total = N * Kp;
    pack_weights<<<(total + 255) / 256, 256, 0, stream>>>(W, dst, N, Kreal, Kp, Cin, total);
  };
  auto gemm = [&](const h16* A, const h16* Wp, const float* bias, h16* outp,
                  int M, int Kp, int N, int ldOut, int act,
                  const float* g, const float* bb, const float* mm, const float* vv) {
    if ((N & 63) == 0) {                          // 16x64 tile / wave
      int waves = (M >> 4) * (N >> 6);
      gemm_wmma<4><<<(waves + 3) / 4, 128, 0, stream>>>(A, Wp, bias, outp, M, Kp, N,
                                                        ldOut, act, g, bb, mm, vv);
    } else {                                      // 16x32 tile / wave
      int waves = (M >> 4) * (N >> 5);
      gemm_wmma<2><<<(waves + 3) / 4, 128, 0, stream>>>(A, Wp, bias, outp, M, Kp, N,
                                                        ldOut, act, g, bb, mm, vv);
    }
  };

  // ---- weight packing (tiny) ----
  pack(c2w,  pw2,   32, 144, 160, 16);
  pack(c3w,  pw3,   64, 288, 288, 32);
  pack(fc1w, pwfc1, 128, 256, 256, 0);
  pack(fc2w, pwfc2, 64, 128, 128, 0);
  pack(q1w,  pwq1,  512, 784, 800, 0);
  pack(q2w,  pwq2,  256, 512, 512, 0);
  pack(f1w,  pwf1,  128,  68,  96, 0);
  pack(f2w,  pwf2,  64, 128, 128, 0);
  pack(f3w,  pwf3,  32,  64,  64, 0);

  // ---- CNN branch ----
  {
    int total = BATCH * 196 * 16;
    conv1_bn_relu<<<(total + 255) / 256, 256, 0, stream>>>(
        x, c1w, c1b, bn1g, bn1b, bn1m, bn1v, h1, total);
  }
  {
    int total = M49 * 160;
    im2col_3x3<<<(total + 255) / 256, 256, 0, stream>>>(h1, A2, 14, 16, 2, 144, 160, total);
  }
  gemm(A2, pw2, c2b, h2, M49, 160, 32, 32, 1, bn2g, bn2b, bn2m, bn2v);
  {
    int total = M49 * 288;
    im2col_3x3<<<(total + 255) / 256, 256, 0, stream>>>(h2, A3, 7, 32, 1, 288, 288, total);
  }
  gemm(A3, pw3, c3b, h3, M49, 288, 64, 64, 1, nullptr, nullptr, nullptr, nullptr);
  {
    int total = BATCH * 256;
    pool7to2<<<(total + 255) / 256, 256, 0, stream>>>(h3, pooled, total);
  }
  gemm(pooled, pwfc1, fc1b, fc1o, BATCH, 256, 128, 128, 1, nullptr, nullptr, nullptr, nullptr);
  gemm(fc1o, pwfc2, fc2b, fcat, BATCH, 128, 64, 96, 1, nullptr, nullptr, nullptr, nullptr);

  // ---- quantum branch ----
  {
    int total = BATCH * 800;
    cast_pad_x<<<(total + 255) / 256, 256, 0, stream>>>(x, xq, total);
  }
  gemm(xq,  pwq1, q1b, q1o, BATCH, 800, 512, 512, 1, nullptr, nullptr, nullptr, nullptr);
  gemm(q1o, pwq2, q2b, q2o, BATCH, 512, 256, 256, 2, nullptr, nullptr, nullptr, nullptr);
  quantum_kernel<<<BATCH / 8, 256, 0, stream>>>(q2o, qwts, fcat, BATCH);

  // ---- fusion head ----
  gemm(fcat, pwf1, f1b, f1o, BATCH, 96, 128, 128, 1, fbn1g, fbn1b, fbn1m, fbn1v);
  gemm(f1o,  pwf2, f2b, f2o, BATCH, 128, 64, 64, 1, fbn2g, fbn2b, fbn2m, fbn2v);
  gemm(f2o,  pwf3, f3b, f3o, BATCH, 64, 32, 32, 1, nullptr, nullptr, nullptr, nullptr);
  f4_sigmoid<<<BATCH / 256, 256, 0, stream>>>(f3o, f4w, f4b, (float*)d_out, BATCH);
}
